// Mamba_sflow_60309930771136
// MI455X (gfx1250) — compile-verified, matrix operand order checked by reading, not run on hardware
//
#include <hip/hip_runtime.h>
#include <hip/hip_bf16.h>

#define Bx 8
#define Sx 2048
#define Hx 512
#define DIx 1024
#define DSx 16
#define DCx 4
#define DRx 32
#define NLx 8
#define NLABx 2

typedef __attribute__((ext_vector_type(16))) __bf16 v16bf;
typedef __attribute__((ext_vector_type(8)))  float  v8f;

// ---------------------------------------------------------------------------
// elementwise f32 -> bf16 cast (weights, once per call)
// ---------------------------------------------------------------------------
__global__ void cast_bf16_kernel(const float* __restrict__ src,
                                 __hip_bfloat16* __restrict__ dst, size_t n) {
  size_t i = (size_t)blockIdx.x * blockDim.x + threadIdx.x;
  size_t stride = (size_t)gridDim.x * blockDim.x;
  for (; i < n; i += stride) dst[i] = __float2bfloat16(src[i]);
}

// ---------------------------------------------------------------------------
// bf16 WMMA GEMM:  C(M,N) = A(M,K) * W(N,K)^T  ; A,W bf16 row-major, C f32
// block = 256 threads = 8 waves in a 4x2 grid; each wave computes a 32x32 C
// tile as 2x2 WMMA tiles (register blocking: 4 loads -> 4 WMMAs per k-step).
// Block tile: 128 (M) x 64 (N). Requires M%128==0, N%64==0, K%32==0.
// ---------------------------------------------------------------------------
__global__ void gemm_bf16_wmma(const __hip_bfloat16* __restrict__ A,
                               const __hip_bfloat16* __restrict__ W,
                               float* __restrict__ C, int M, int N, int K) {
  const int lane = threadIdx.x & 31;
  const int wave = threadIdx.x >> 5;           // 0..7
  const int wm = wave >> 1;                    // 0..3
  const int wn = wave & 1;                     // 0..1
  const int m0 = blockIdx.y * 128 + wm * 32;   // wave's 32-row strip
  const int n0 = blockIdx.x * 64 + wn * 32;    // wave's 32-col strip
  const int row  = lane & 15;
  const int half = lane >> 4;

  const __hip_bfloat16* A0p = A + (size_t)(m0 + row) * K + half * 16;
  const __hip_bfloat16* A1p = A0p + (size_t)16 * K;
  const __hip_bfloat16* W0p = W + (size_t)(n0 + row) * K + half * 16;
  const __hip_bfloat16* W1p = W0p + (size_t)16 * K;

  v8f acc00 = {}, acc01 = {}, acc10 = {}, acc11 = {};
  for (int k0 = 0; k0 < K; k0 += 32) {
    v16bf a0 = *reinterpret_cast<const v16bf*>(A0p + k0);
    v16bf a1 = *reinterpret_cast<const v16bf*>(A1p + k0);
    v16bf b0 = *reinterpret_cast<const v16bf*>(W0p + k0);
    v16bf b1 = *reinterpret_cast<const v16bf*>(W1p + k0);
    // prefetch next K-step (lowers to global_prefetch_b8 on gfx1250)
    if (k0 + 32 < K) {
      __builtin_prefetch(A0p + k0 + 32, 0, 1);
      __builtin_prefetch(A1p + k0 + 32, 0, 1);
      __builtin_prefetch(W0p + k0 + 32, 0, 1);
      __builtin_prefetch(W1p + k0 + 32, 0, 1);
    }
    acc00 = __builtin_amdgcn_wmma_f32_16x16x32_bf16(false, a0, false, b0,
                                                    (short)0, acc00, false, false);
    acc01 = __builtin_amdgcn_wmma_f32_16x16x32_bf16(false, a0, false, b1,
                                                    (short)0, acc01, false, false);
    acc10 = __builtin_amdgcn_wmma_f32_16x16x32_bf16(false, a1, false, b0,
                                                    (short)0, acc10, false, false);
    acc11 = __builtin_amdgcn_wmma_f32_16x16x32_bf16(false, a1, false, b1,
                                                    (short)0, acc11, false, false);
  }
  // C/D layout: VGPR r, lanes 0-15 -> M=r ; lanes 16-31 -> M=r+8 ; N = lane&15
  const int cn = lane & 15;
  float* C00 = C + (size_t)(m0 + half * 8) * N + n0 + cn;
  float* C01 = C00 + 16;
  float* C10 = C00 + (size_t)16 * N;
  float* C11 = C10 + 16;
#pragma unroll
  for (int r = 0; r < 8; ++r) {
    C00[(size_t)r * N] = acc00[r];
    C01[(size_t)r * N] = acc01[r];
    C10[(size_t)r * N] = acc10[r];
    C11[(size_t)r * N] = acc11[r];
  }
}

// ---------------------------------------------------------------------------
// embedding + LN + gate   (one token per block, 256 threads, 2 elems/thread)
// ---------------------------------------------------------------------------
__global__ void embed_ln_kernel(const int* __restrict__ ids,
                                const float* __restrict__ gate,
                                const float* __restrict__ word_emb,
                                const float* __restrict__ pos_emb2,
                                const float* __restrict__ pe,
                                const float* __restrict__ g,
                                const float* __restrict__ bparm,
                                float* __restrict__ seq,
                                __hip_bfloat16* __restrict__ seq_bf) {
  const int tok = blockIdx.x;       // b*S + s
  const int s = tok % Sx;
  const int id = ids[tok];
  const int t = threadIdx.x;
  __shared__ float red[256];
  __shared__ float smu, srs;

  float e[2];
#pragma unroll
  for (int i = 0; i < 2; ++i) {
    int h = t + i * 256;
    e[i] = pos_emb2[s * Hx + h] * word_emb[(size_t)id * Hx + h] + pe[s * Hx + h];
  }
  red[t] = e[0] + e[1];
  __syncthreads();
  for (int off = 128; off > 0; off >>= 1) {
    if (t < off) red[t] += red[t + off];
    __syncthreads();
  }
  if (t == 0) smu = red[0] / (float)Hx;
  __syncthreads();
  float mu = smu;
  red[t] = (e[0] - mu) * (e[0] - mu) + (e[1] - mu) * (e[1] - mu);
  __syncthreads();
  for (int off = 128; off > 0; off >>= 1) {
    if (t < off) red[t] += red[t + off];
    __syncthreads();
  }
  if (t == 0) srs = rsqrtf(red[0] / (float)Hx + 1e-12f);
  __syncthreads();
  float rs = srs;
  float gv = gate[tok];
#pragma unroll
  for (int i = 0; i < 2; ++i) {
    int h = t + i * 256;
    float o = (g[h] * (e[i] - mu) * rs + bparm[h]) * gv;
    seq[(size_t)tok * Hx + h] = o;
    seq_bf[(size_t)tok * Hx + h] = __float2bfloat16(o);
  }
}

// ---------------------------------------------------------------------------
// residual + LN + gate (same shape as embed_ln)
// ---------------------------------------------------------------------------
__global__ void resid_ln_gate_kernel(const float* __restrict__ hout,
                                     const float* __restrict__ seqin,
                                     const float* __restrict__ gate,
                                     const float* __restrict__ g,
                                     const float* __restrict__ bparm,
                                     float* __restrict__ seqout,
                                     __hip_bfloat16* __restrict__ seq_bf) {
  const int tok = blockIdx.x;
  const int t = threadIdx.x;
  __shared__ float red[256];
  __shared__ float smu, srs;

  float e[2];
#pragma unroll
  for (int i = 0; i < 2; ++i) {
    int h = t + i * 256;
    e[i] = hout[(size_t)tok * Hx + h] + seqin[(size_t)tok * Hx + h];
  }
  red[t] = e[0] + e[1];
  __syncthreads();
  for (int off = 128; off > 0; off >>= 1) {
    if (t < off) red[t] += red[t + off];
    __syncthreads();
  }
  if (t == 0) smu = red[0] / (float)Hx;
  __syncthreads();
  float mu = smu;
  red[t] = (e[0] - mu) * (e[0] - mu) + (e[1] - mu) * (e[1] - mu);
  __syncthreads();
  for (int off = 128; off > 0; off >>= 1) {
    if (t < off) red[t] += red[t + off];
    __syncthreads();
  }
  if (t == 0) srs = rsqrtf(red[0] / (float)Hx + 1e-12f);
  __syncthreads();
  float rs = srs;
  float gv = gate[tok];
#pragma unroll
  for (int i = 0; i < 2; ++i) {
    int h = t + i * 256;
    float o = (g[h] * (e[i] - mu) * rs + bparm[h]) * gv;
    seqout[(size_t)tok * Hx + h] = o;
    seq_bf[(size_t)tok * Hx + h] = __float2bfloat16(o);
  }
}

// ---------------------------------------------------------------------------
// causal depthwise conv (DC=4) + SiLU ; reads x half of xz
// ---------------------------------------------------------------------------
__global__ void conv_silu_kernel(const float* __restrict__ xz,
                                 const float* __restrict__ cw,
                                 const float* __restrict__ cb,
                                 float* __restrict__ xact,
                                 __hip_bfloat16* __restrict__ xbf) {
  size_t idx = (size_t)blockIdx.x * blockDim.x + threadIdx.x;  // B*S*DI
  const int d = idx % DIx;
  size_t bt = idx / DIx;
  const int t = bt % Sx;
  const size_t b = bt / Sx;
  float acc = cb[d];
#pragma unroll
  for (int k = 0; k < DCx; ++k) {
    int tt = t + k - (DCx - 1);
    if (tt >= 0)
      acc += cw[d * DCx + k] * xz[((b * Sx + tt)) * (size_t)(2 * DIx) + d];
  }
  float s = acc / (1.0f + __expf(-acc));  // silu
  xact[idx] = s;
  xbf[idx] = __float2bfloat16(s);
}

// ---------------------------------------------------------------------------
// dt = softplus(xdb[:, :DR] @ Wdt^T + bdt) ; one token per block
// ---------------------------------------------------------------------------
__global__ void dt_kernel(const float* __restrict__ xdb,
                          const float* __restrict__ Wdt,
                          const float* __restrict__ bdt,
                          float* __restrict__ dtb) {
  const size_t bt = blockIdx.x;
  __shared__ float sx[DRx];
  if (threadIdx.x < DRx) sx[threadIdx.x] = xdb[bt * 64 + threadIdx.x];
  __syncthreads();
#pragma unroll
  for (int i = 0; i < 4; ++i) {
    int d = threadIdx.x + i * 256;
    float acc = bdt[d];
#pragma unroll
    for (int r = 0; r < DRx; ++r) acc += sx[r] * Wdt[d * DRx + r];
    float sp = (acc > 20.0f) ? acc : log1pf(__expf(acc));
    dtb[bt * DIx + d] = sp;
  }
}

// ---------------------------------------------------------------------------
// selective scan: one block per batch, thread = channel d, state in registers
// ---------------------------------------------------------------------------
__global__ void scan_kernel(const float* __restrict__ xdb,
                            const float* __restrict__ dtb,
                            const float* __restrict__ xact,
                            const float* __restrict__ Alog,
                            const float* __restrict__ Dp,
                            float* __restrict__ ybuf) {
  const int b = blockIdx.x;
  const int d = threadIdx.x;  // 0..DI-1
  float A[DSx], h[DSx];
#pragma unroll
  for (int n = 0; n < DSx; ++n) {
    A[n] = -__expf(Alog[d * DSx + n]);
    h[n] = 0.0f;
  }
  const float dval = Dp[d];
  __shared__ float sB[DSx];
  __shared__ float sC[DSx];
  for (int t = 0; t < Sx; ++t) {
    size_t bt = (size_t)b * Sx + t;
    if (d < 32) {
      float v = xdb[bt * 64 + 32 + d];
      if (d < DSx) sB[d] = v; else sC[d - DSx] = v;
    }
    __syncthreads();
    float dtv = dtb[bt * DIx + d];
    float xv = xact[bt * DIx + d];
    float dtx = dtv * xv;
    float y = 0.0f;
#pragma unroll
    for (int n = 0; n < DSx; ++n) {
      h[n] = __expf(dtv * A[n]) * h[n] + dtx * sB[n];
      y += h[n] * sC[n];
    }
    ybuf[bt * DIx + d] = y + dval * xv;
    __syncthreads();
  }
}

// ---------------------------------------------------------------------------
// y *= silu(z) ; -> bf16
// ---------------------------------------------------------------------------
__global__ void gate_silu_kernel(const float* __restrict__ ybuf,
                                 const float* __restrict__ xz,
                                 __hip_bfloat16* __restrict__ ybf) {
  size_t idx = (size_t)blockIdx.x * blockDim.x + threadIdx.x;  // B*S*DI
  const int d = idx % DIx;
  size_t bt = idx / DIx;
  float z = xz[bt * (size_t)(2 * DIx) + DIx + d];
  float s = z / (1.0f + __expf(-z));
  ybf[idx] = __float2bfloat16(ybuf[idx] * s);
}

// ---------------------------------------------------------------------------
// gated max pool over sequence ; block = batch, thread = h
// ---------------------------------------------------------------------------
__global__ void pool_kernel(const float* __restrict__ seq,
                            const float* __restrict__ gate,
                            float* __restrict__ pooled) {
  const int b = blockIdx.x;
  const int h = threadIdx.x;  // 0..H-1
  float m = -3.4e38f;
  for (int t = 0; t < Sx; ++t) {
    float v = seq[((size_t)b * Sx + t) * Hx + h] * gate[b * Sx + t];
    m = fmaxf(m, v);
  }
  pooled[b * Hx + h] = m;
}

__global__ void copy_kernel(const float* __restrict__ src, float* __restrict__ dst,
                            size_t n) {
  size_t i = (size_t)blockIdx.x * blockDim.x + threadIdx.x;
  if (i < n) dst[i] = src[i];
}

// ---------------------------------------------------------------------------
// head: dense+GELU, ELU style, 4-group GroupNorm, affine mix, logits
// block = batch row, 512 threads (thread = feature j)
// ---------------------------------------------------------------------------
__global__ void head_kernel(const float* __restrict__ pooled,
                            const float* __restrict__ age_sex,
                            const float* __restrict__ d2w,
                            const float* __restrict__ d2b,
                            const float* __restrict__ mw,
                            const float* __restrict__ mb,
                            const float* __restrict__ gng,
                            const float* __restrict__ gnb,
                            const float* __restrict__ hw,
                            const float* __restrict__ hb,
                            float* __restrict__ out_logits,
                            float* __restrict__ out_feature) {
  const int b = blockIdx.x;
  const int j = threadIdx.x;  // 0..511
  __shared__ float sp[Hx];
  __shared__ float red[Hx];
  __shared__ float gmu[4], grs[4];

  sp[j] = pooled[b * Hx + j];
  __syncthreads();

  float acc = d2b[j];
#pragma unroll 8
  for (int h = 0; h < Hx; ++h) acc += sp[h] * d2w[j * Hx + h];
  // gelu_new
  const float c = 0.7978845608028654f;
  float x3 = acc * acc * acc;
  float feat = 0.5f * acc * (1.0f + tanhf(c * (acc + 0.044715f * x3)));

  float a0 = age_sex[b * 2], a1 = age_sex[b * 2 + 1];
  float s0 = a0 * mw[j * 2] + a1 * mw[j * 2 + 1] + mb[j];
  float s1 = a0 * mw[(j + Hx) * 2] + a1 * mw[(j + Hx) * 2 + 1] + mb[j + Hx];
  s0 = (s0 > 0.0f) ? s0 : (__expf(s0) - 1.0f);
  s1 = (s1 > 0.0f) ? s1 : (__expf(s1) - 1.0f);

  // group norm: 4 groups of 128
  const int g = j >> 7;
  const int lj = j & 127;
  red[j] = feat;
  __syncthreads();
  for (int off = 64; off > 0; off >>= 1) {
    if (lj < off) red[j] += red[j + off];
    __syncthreads();
  }
  if (lj == 0) gmu[g] = red[g << 7] / 128.0f;
  __syncthreads();
  float mu = gmu[g];
  float dv = feat - mu;
  red[j] = dv * dv;
  __syncthreads();
  for (int off = 64; off > 0; off >>= 1) {
    if (lj < off) red[j] += red[j + off];
    __syncthreads();
  }
  if (lj == 0) grs[g] = rsqrtf(red[g << 7] / 128.0f + 1e-5f);
  __syncthreads();
  float gn = dv * grs[g] * gng[j] + gnb[j];
  float feature = (1.0f + s0) * gn + s1;
  out_feature[b * Hx + j] = feature;

  // logits: 2 rows of head_w
  red[j] = feature * hw[j];
  __syncthreads();
  for (int off = 256; off > 0; off >>= 1) {
    if (j < off) red[j] += red[j + off];
    __syncthreads();
  }
  if (j == 0) out_logits[b * NLABx + 0] = red[0] + hb[0];
  __syncthreads();
  red[j] = feature * hw[Hx + j];
  __syncthreads();
  for (int off = 256; off > 0; off >>= 1) {
    if (j < off) red[j] += red[j + off];
    __syncthreads();
  }
  if (j == 0) out_logits[b * NLABx + 1] = red[0] + hb[1];
}

// ---------------------------------------------------------------------------
extern "C" void kernel_launch(void* const* d_in, const int* in_sizes, int n_in,
                              void* d_out, int out_size, void* d_ws, size_t ws_size,
                              hipStream_t stream) {
  (void)in_sizes; (void)n_in; (void)out_size; (void)ws_size;

  const int*   input_ids = (const int*)d_in[0];
  const float* gate      = (const float*)d_in[1];
  const float* age_sex   = (const float*)d_in[2];
  const float* word_emb  = (const float*)d_in[3];
  const float* pos_emb2  = (const float*)d_in[4];
  const float* pe        = (const float*)d_in[5];
  const float* emb_ln_g  = (const float*)d_in[6];
  const float* emb_ln_b  = (const float*)d_in[7];
  const float* in_proj_w = (const float*)d_in[8];
  const float* conv_w    = (const float*)d_in[9];
  const float* conv_b    = (const float*)d_in[10];
  const float* x_proj_w  = (const float*)d_in[11];
  const float* dt_proj_w = (const float*)d_in[12];
  const float* dt_proj_b = (const float*)d_in[13];
  const float* A_log     = (const float*)d_in[14];
  const float* D_param   = (const float*)d_in[15];
  const float* out_proj_w= (const float*)d_in[16];
  const float* blk_ln_g  = (const float*)d_in[17];
  const float* blk_ln_b  = (const float*)d_in[18];
  const float* dense2_w  = (const float*)d_in[19];
  const float* dense2_b  = (const float*)d_in[20];
  const float* male_fc_w = (const float*)d_in[21];
  const float* male_fc_b = (const float*)d_in[22];
  const float* gn_g      = (const float*)d_in[23];
  const float* gn_b      = (const float*)d_in[24];
  const float* head_w    = (const float*)d_in[25];
  const float* head_b    = (const float*)d_in[26];

  const size_t BS = (size_t)Bx * Sx;  // 16384

  // -------- workspace carve-out --------
  char* w = (char*)d_ws;
  auto carve = [&](size_t bytes) -> void* {
    void* p = (void*)w;
    w += (bytes + 255) & ~(size_t)255;
    return p;
  };
  const size_t n_wi = (size_t)NLx * 2 * DIx * Hx;
  const size_t n_wx = (size_t)NLx * 64 * DIx;
  const size_t n_wo = (size_t)NLx * Hx * DIx;

  __hip_bfloat16* wbf_in  = (__hip_bfloat16*)carve(n_wi * 2);
  __hip_bfloat16* wbf_x   = (__hip_bfloat16*)carve(n_wx * 2);
  __hip_bfloat16* wbf_out = (__hip_bfloat16*)carve(n_wo * 2);
  float*          seq     = (float*)carve(BS * Hx * 4);
  __hip_bfloat16* seq_bf  = (__hip_bfloat16*)carve(BS * Hx * 2);
  float*          xz      = (float*)carve(BS * 2 * DIx * 4);
  float*          xact    = (float*)carve(BS * DIx * 4);
  __hip_bfloat16* xact_bf = (__hip_bfloat16*)carve(BS * DIx * 2);
  float*          xdb     = (float*)carve(BS * 64 * 4);
  float*          dtb     = (float*)carve(BS * DIx * 4);
  float*          ybuf    = (float*)carve(BS * DIx * 4);
  __hip_bfloat16* ybf     = (__hip_bfloat16*)carve(BS * DIx * 2);
  float*          hout    = (float*)carve(BS * Hx * 4);
  float*          pooled  = (float*)carve((size_t)Bx * Hx * 4);

  // -------- weight casts (deterministic, idempotent) --------
  cast_bf16_kernel<<<2048, 256, 0, stream>>>(in_proj_w, wbf_in, n_wi);
  cast_bf16_kernel<<<512, 256, 0, stream>>>(x_proj_w, wbf_x, n_wx);
  cast_bf16_kernel<<<2048, 256, 0, stream>>>(out_proj_w, wbf_out, n_wo);

  // -------- embedding + LN + gate --------
  embed_ln_kernel<<<(int)BS, 256, 0, stream>>>(input_ids, gate, word_emb,
                                               pos_emb2, pe, emb_ln_g, emb_ln_b,
                                               seq, seq_bf);

  const size_t nBSDI = BS * DIx;
  const int ew_blocks = (int)(nBSDI / 256);

  // -------- layers --------
  for (int l = 0; l < NLx; ++l) {
    const __hip_bfloat16* Wi = wbf_in + (size_t)l * 2 * DIx * Hx;
    const __hip_bfloat16* Wx = wbf_x + (size_t)l * 64 * DIx;
    const __hip_bfloat16* Wo = wbf_out + (size_t)l * Hx * DIx;

    // xz = seq @ Wi^T      (M=BS, N=2*DI, K=H)
    gemm_bf16_wmma<<<dim3(2 * DIx / 64, (int)(BS / 128)), 256, 0, stream>>>(
        seq_bf, Wi, xz, (int)BS, 2 * DIx, Hx);

    // causal conv + silu
    conv_silu_kernel<<<ew_blocks, 256, 0, stream>>>(
        xz, conv_w + (size_t)l * DIx * DCx, conv_b + (size_t)l * DIx, xact, xact_bf);

    // xdb = x @ Wx^T       (M=BS, N=64, K=DI)
    gemm_bf16_wmma<<<dim3(64 / 64, (int)(BS / 128)), 256, 0, stream>>>(
        xact_bf, Wx, xdb, (int)BS, 64, DIx);

    // dt = softplus(xdb[:, :DR] @ Wdt^T + bdt)
    dt_kernel<<<(int)BS, 256, 0, stream>>>(
        xdb, dt_proj_w + (size_t)l * DIx * DRx, dt_proj_b + (size_t)l * DIx, dtb);

    // selective scan
    scan_kernel<<<Bx, DIx, 0, stream>>>(
        xdb, dtb, xact, A_log + (size_t)l * DIx * DSx, D_param + (size_t)l * DIx,
        ybuf);

    // y *= silu(z)
    gate_silu_kernel<<<ew_blocks, 256, 0, stream>>>(ybuf, xz, ybf);

    // hout = y @ Wo^T      (M=BS, N=H, K=DI)
    gemm_bf16_wmma<<<dim3(Hx / 64, (int)(BS / 128)), 256, 0, stream>>>(
        ybf, Wo, hout, (int)BS, Hx, DIx);

    // seq = gate * LN(hout + seq)
    resid_ln_gate_kernel<<<(int)BS, 256, 0, stream>>>(
        hout, seq, gate, blk_ln_g + (size_t)l * Hx, blk_ln_b + (size_t)l * Hx,
        seq, seq_bf);
  }

  // -------- outputs --------
  float* out = (float*)d_out;
  float* out_logits  = out;                       // B*NLAB = 16
  float* out_gate    = out + Bx * NLABx;          // B*S
  float* out_feature = out + Bx * NLABx + BS;     // B*H

  pool_kernel<<<Bx, Hx, 0, stream>>>(seq, gate, pooled);
  copy_kernel<<<(int)(BS / 256), 256, 0, stream>>>(gate, out_gate, BS);
  head_kernel<<<Bx, Hx, 0, stream>>>(pooled, age_sex, dense2_w, dense2_b,
                                     male_fc_w, male_fc_b, gn_g, gn_b, head_w,
                                     head_b, out_logits, out_feature);
}
